// MoETransformerBlock_39694087749957
// MI455X (gfx1250) — matrix operand order repaired and checked by reading, not compiled, not run on hardware
//
#include <hip/hip_runtime.h>

#define DEV __device__ __forceinline__

typedef __bf16 v16bf __attribute__((ext_vector_type(16)));
typedef float  v8f   __attribute__((ext_vector_type(8)));
typedef int    v4i   __attribute__((ext_vector_type(4)));
typedef unsigned short u16;

// Problem constants (fixed by the reference)
constexpr int Bc = 2, Sc = 2048, Dc = 1024, Hc = 16, HDc = 64, FFc = 4096, Ec = 8;
constexpr int Nc  = Bc * Sc;   // 4096 tokens
constexpr int D3c = 3 * Dc;    // 3072
constexpr int NAc = Nc * 2;    // 8192 assignments (top-2)

DEV u16 f2bf(float f) {
  unsigned u = __builtin_bit_cast(unsigned, f);
  unsigned r = u + 0x7FFFu + ((u >> 16) & 1u);
  return (u16)(r >> 16);
}

DEV v8f zero8() {
  v8f z;
#pragma unroll
  for (int i = 0; i < 8; ++i) z[i] = 0.f;
  return z;
}

// Assemble a 16-element bf16 fragment from two 16-byte LDS chunks.
DEV v16bf ld_frag(const u16* p0, const u16* p1) {
  union { v16bf v; v4i q[2]; } u;
  u.q[0] = *(const v4i*)p0;
  u.q[1] = *(const v4i*)p1;
  return u.v;
}

DEV v8f wmma_bf16(v16bf a, v16bf b, v8f c) {
  return __builtin_amdgcn_wmma_f32_16x16x32_bf16(false, a, false, b, (short)0, c,
                                                 false, false);
}

// CDNA5 async global->LDS copy of 32 bytes (two B128 ops; IOFFSET applies to
// both the LDS and the global address per ISA 15.18.3). Tracked by ASYNCcnt.
DEV void async_cp32(unsigned ldsaddr, const u16* gaddr) {
  asm volatile(
      "global_load_async_to_lds_b128 %0, %1, off\n\t"
      "global_load_async_to_lds_b128 %0, %1, off offset:16"
      :
      : "v"(ldsaddr), "v"(gaddr)
      : "memory");
}

DEV void wait_async0() { asm volatile("s_wait_asynccnt 0x0" ::: "memory"); }

// Low 32 bits of a flat pointer to LDS == workgroup-relative LDS byte offset.
DEV unsigned lds_addr(const void* p) {
  return (unsigned)(unsigned long long)p;
}

// ---------------------------------------------------------------- f32 -> bf16
__global__ void k_cvt(const float* __restrict__ in, u16* __restrict__ out, long n) {
  long i = (long)blockIdx.x * blockDim.x + threadIdx.x;
  long st = (long)gridDim.x * blockDim.x;
  for (; i < n; i += st) out[i] = f2bf(in[i]);
}

__global__ void k_zero(int* p, int n) {
  int i = blockIdx.x * blockDim.x + threadIdx.x;
  if (i < n) p[i] = 0;
}

// -------------------------------------------------------------------- GEMM
// C[M,N] = A[M,Kd] @ B[N,Kd]^T (+bias) (+relu). Optional expert mode:
// cnts/offs give per-expert row segment; row_map gathers A rows (tokens).
// Tiles stream into double-buffered LDS via async global->LDS copies.
__global__ __launch_bounds__(256) void k_gemm(
    const u16* __restrict__ A, const u16* __restrict__ Bw,
    const float* __restrict__ bias, long biasExpStride,
    float* __restrict__ Cf, u16* __restrict__ Cb,
    int M, int N, int Kd, int relu,
    const int* __restrict__ row_map,
    const int* __restrict__ cnts, const int* __restrict__ offs,
    long strideBexp, const u16* __restrict__ zeros) {
  constexpr int LDA = 40;  // 32 + 8 pad (80B row stride, 16B aligned)
  int e = blockIdx.z;
  int Me = M, off = 0;
  if (cnts) { Me = cnts[e]; off = offs[e]; }
  int m0 = blockIdx.x * 128;
  if (m0 >= Me) return;
  int n0 = blockIdx.y * 128;
  const u16* Bp = Bw + (long)e * strideBexp;
  const float* bp = bias ? (bias + (long)e * biasExpStride) : nullptr;

  __shared__ __align__(16) u16 As[2][128 * LDA];
  __shared__ __align__(16) u16 Bs[2][128 * LDA];

  int tid = threadIdx.x;
  int lane = tid & 31, wid = tid >> 5;
  int wm = wid & 3, wn = wid >> 2;  // waves: 4 (M) x 2 (N); wave tile 32x64
  int hl = lane >> 4, l16 = lane & 15;

  v8f acc[2][4];
#pragma unroll
  for (int i = 0; i < 2; ++i)
#pragma unroll
    for (int j = 0; j < 4; ++j) acc[i][j] = zero8();

  // cooperative copy: each thread owns one 32B chunk of A and one of B
  int arow = tid >> 1;
  int akb = (tid & 1) * 16;
  int r_local = m0 + arow;
  bool a_ok = (r_local < Me);
  int tokenIdx = 0;
  if (a_ok) tokenIdx = row_map ? row_map[off + r_local] : (off + r_local);
  const u16* a_src = a_ok ? (A + (long)tokenIdx * Kd + akb) : zeros;
  const u16* b_src = Bp + (long)(n0 + arow) * Kd + akb;
  unsigned aA[2], aB[2];
  aA[0] = lds_addr(&As[0][arow * LDA + akb]);
  aA[1] = lds_addr(&As[1][arow * LDA + akb]);
  aB[0] = lds_addr(&Bs[0][arow * LDA + akb]);
  aB[1] = lds_addr(&Bs[1][arow * LDA + akb]);

  // prologue: tile 0 in flight
  async_cp32(aA[0], a_src);
  async_cp32(aB[0], b_src);
  if (a_ok) a_src += 32;
  b_src += 32;

  int ksteps = Kd >> 5;
  for (int ks = 0; ks < ksteps; ++ks) {
    wait_async0();
    __syncthreads();  // tile ks resident; everyone done reading buffer ks^1
    int cur = ks & 1;
    if (ks + 1 < ksteps) {
      async_cp32(aA[cur ^ 1], a_src);
      async_cp32(aB[cur ^ 1], b_src);
      if (a_ok) a_src += 32;
      b_src += 32;
    }

    const u16* Ab = As[cur];
    const u16* Bb = Bs[cur];
    v16bf afr[2], bfr[4];
#pragma unroll
    for (int mi = 0; mi < 2; ++mi) {
      const u16* p = Ab + (wm * 32 + mi * 16 + l16) * LDA + hl * 8;
      afr[mi] = ld_frag(p, p + 16);
    }
#pragma unroll
    for (int ni = 0; ni < 4; ++ni) {
      const u16* p = Bb + (wn * 64 + ni * 16 + l16) * LDA + hl * 8;
      bfr[ni] = ld_frag(p, p + 16);
    }
#pragma unroll
    for (int mi = 0; mi < 2; ++mi)
#pragma unroll
      for (int ni = 0; ni < 4; ++ni)
        acc[mi][ni] = wmma_bf16(afr[mi], bfr[ni], acc[mi][ni]);
  }

#pragma unroll
  for (int mi = 0; mi < 2; ++mi) {
#pragma unroll
    for (int ni = 0; ni < 4; ++ni) {
      int colg = n0 + wn * 64 + ni * 16 + l16;
      float bv = bp ? bp[colg] : 0.f;
#pragma unroll
      for (int r = 0; r < 8; ++r) {
        int ml = m0 + wm * 32 + mi * 16 + hl * 8 + r;
        if (ml < Me) {
          float v = acc[mi][ni][r] + bv;
          if (relu) v = fmaxf(v, 0.f);
          long ci = (long)(off + ml) * N + colg;
          if (Cf) Cf[ci] = v;
          if (Cb) Cb[ci] = f2bf(v);
        }
      }
    }
  }
}

// ---------------------------------------------------------- flash attention
__global__ __launch_bounds__(256) void k_attn(const u16* __restrict__ qkv,
                                              u16* __restrict__ ctxb) {
  const int b = blockIdx.z, h = blockIdx.y, qt = blockIdx.x;
  const int tid = threadIdx.x, lane = tid & 31, wid = tid >> 5;
  const int hl = lane >> 4, l16 = lane & 15;
  const long rowbase = (long)b * Sc + (long)qt * 128;
  const long qoff = (long)h * HDc;
  const long koff = qoff + Dc;
  const long voff = qoff + 2 * Dc;

  __shared__ __align__(16) u16 Qs[128 * 72];
  __shared__ __align__(16) u16 Ks[32 * 72];
  __shared__ __align__(16) u16 Vt[64 * 40];
  __shared__ __align__(16) u16 Pw[8 * 16 * 40];

  // Load Q tile [128][64]
  for (int c = tid; c < 1024; c += 256) {
    int row = c >> 3, db = (c & 7) * 8;
    const u16* src = qkv + (rowbase + row) * (long)D3c + qoff + db;
    *(v4i*)(Qs + row * 72 + db) = *(const v4i*)src;
  }

  v8f cacc[4];
#pragma unroll
  for (int i = 0; i < 4; ++i) cacc[i] = zero8();
  float m8[8], l8[8];
#pragma unroll
  for (int r = 0; r < 8; ++r) { m8[r] = -1e30f; l8[r] = 0.f; }

  u16* myP = Pw + wid * (16 * 40);

  for (int kv0 = 0; kv0 < Sc; kv0 += 32) {
    __syncthreads();
    {  // K chunk [32][64] row-major; V chunk transposed [64][32]
      int row = tid >> 3, db = (tid & 7) * 8;
      const u16* ksrc = qkv + ((long)b * Sc + kv0 + row) * D3c + koff + db;
      *(v4i*)(Ks + row * 72 + db) = *(const v4i*)ksrc;
      const u16* vsrc = qkv + ((long)b * Sc + kv0 + row) * D3c + voff + db;
      u16 tmp[8];
      *(v4i*)tmp = *(const v4i*)vsrc;
#pragma unroll
      for (int j = 0; j < 8; ++j) Vt[(db + j) * 40 + row] = tmp[j];
    }
    __syncthreads();

    // scores: 16 q-rows (this wave) x 32 kv cols, K-dim = 64
    v8f s0 = zero8(), s1 = zero8();
#pragma unroll
    for (int kstep = 0; kstep < 64; kstep += 32) {
      const u16* ap = Qs + (wid * 16 + l16) * 72 + kstep + hl * 8;
      v16bf a = ld_frag(ap, ap + 16);
      const u16* bp0 = Ks + l16 * 72 + kstep + hl * 8;
      const u16* bp1 = Ks + (16 + l16) * 72 + kstep + hl * 8;
      s0 = wmma_bf16(a, ld_frag(bp0, bp0 + 16), s0);
      s1 = wmma_bf16(a, ld_frag(bp1, bp1 + 16), s1);
    }

    // online softmax update (row stats replicated per half-wave)
#pragma unroll
    for (int r = 0; r < 8; ++r) {
      float x0 = s0[r] * 0.125f, x1v = s1[r] * 0.125f;
      float mr = fmaxf(x0, x1v);
      mr = fmaxf(mr, __shfl_xor(mr, 1, 32));
      mr = fmaxf(mr, __shfl_xor(mr, 2, 32));
      mr = fmaxf(mr, __shfl_xor(mr, 4, 32));
      mr = fmaxf(mr, __shfl_xor(mr, 8, 32));
      float nm = fmaxf(m8[r], mr);
      float sc = __expf(m8[r] - nm);
      float p0 = __expf(x0 - nm);
      float p1 = __expf(x1v - nm);
      float rs = p0 + p1;
      rs += __shfl_xor(rs, 1, 32);
      rs += __shfl_xor(rs, 2, 32);
      rs += __shfl_xor(rs, 4, 32);
      rs += __shfl_xor(rs, 8, 32);
      l8[r] = l8[r] * sc + rs;
      m8[r] = nm;
#pragma unroll
      for (int ni = 0; ni < 4; ++ni) cacc[ni][r] *= sc;
      int prow = hl * 8 + r;  // C-layout -> LDS (A-layout transpose)
      myP[prow * 40 + l16] = f2bf(p0);
      myP[prow * 40 + 16 + l16] = f2bf(p1);
    }
    asm volatile("s_wait_dscnt 0" ::: "memory");

    // ctx += P @ V
    {
      const u16* ap = myP + l16 * 40 + hl * 8;
      v16bf a = ld_frag(ap, ap + 16);
#pragma unroll
      for (int ni = 0; ni < 4; ++ni) {
        const u16* bp = Vt + (ni * 16 + l16) * 40 + hl * 8;
        cacc[ni] = wmma_bf16(a, ld_frag(bp, bp + 16), cacc[ni]);
      }
    }
  }

#pragma unroll
  for (int r = 0; r < 8; ++r) {
    float inv = 1.f / l8[r];
    long orow = rowbase + wid * 16 + hl * 8 + r;
#pragma unroll
    for (int ni = 0; ni < 4; ++ni)
      ctxb[orow * Dc + qoff + ni * 16 + l16] = f2bf(cacc[ni][r] * inv);
  }
}

// ----------------------------------------------------------- reductions/LN
DEV float block_sum(float v, float* red) {
  int lane = threadIdx.x & 31, wid = threadIdx.x >> 5;
#pragma unroll
  for (int m = 16; m >= 1; m >>= 1) v += __shfl_xor(v, m, 32);
  if (lane == 0) red[wid] = v;
  __syncthreads();
  if (wid == 0) {
    float r = (lane < 8) ? red[lane] : 0.f;
#pragma unroll
    for (int m = 4; m >= 1; m >>= 1) r += __shfl_xor(r, m, 32);
    if (lane == 0) red[0] = r;
  }
  __syncthreads();
  return red[0];
}

__global__ __launch_bounds__(256) void k_ln1(
    const float* __restrict__ x, const float* __restrict__ attn,
    const float* __restrict__ g, const float* __restrict__ be,
    float* __restrict__ x1f, u16* __restrict__ x1b) {
  int n = blockIdx.x, tid = threadIdx.x;
  __shared__ float red1[8], red2[8];
  const float* xr = x + (long)n * Dc;
  const float* ar = attn + (long)n * Dc;
  float t[4], s = 0.f, ss = 0.f;
#pragma unroll
  for (int i = 0; i < 4; ++i) {
    int idx = tid + i * 256;
    float v = xr[idx] + ar[idx];
    t[i] = v; s += v; ss += v * v;
  }
  s = block_sum(s, red1);
  ss = block_sum(ss, red2);
  float mean = s * (1.f / Dc);
  float var = ss * (1.f / Dc) - mean * mean;
  float inv = rsqrtf(var + 1e-5f);
#pragma unroll
  for (int i = 0; i < 4; ++i) {
    int idx = tid + i * 256;
    float v = (t[i] - mean) * inv * g[idx] + be[idx];
    x1f[(long)n * Dc + idx] = v;
    x1b[(long)n * Dc + idx] = f2bf(v);
  }
}

__global__ __launch_bounds__(256) void k_gate(const float* __restrict__ x1,
                                              const float* __restrict__ gw,
                                              const float* __restrict__ gb,
                                              float* __restrict__ logits) {
  int n = blockIdx.x;
  int lane = threadIdx.x & 31, wid = threadIdx.x >> 5;  // wid == expert
  const float* xr = x1 + (long)n * Dc;
  const float* wr = gw + (long)wid * Dc;
  float s = 0.f;
  for (int d = lane; d < Dc; d += 32) s += xr[d] * wr[d];
#pragma unroll
  for (int m = 16; m >= 1; m >>= 1) s += __shfl_xor(s, m, 32);
  if (lane == 0) logits[n * Ec + wid] = s + gb[wid];
}

DEV void top2(const float* l, int& e0, int& e1, float& w0, float& w1) {
  int a = 0; float va = l[0];
#pragma unroll
  for (int e = 1; e < Ec; ++e)
    if (l[e] > va) { va = l[e]; a = e; }
  int bi = -1; float vb = -1e30f;
#pragma unroll
  for (int e = 0; e < Ec; ++e)
    if (e != a && l[e] > vb) { vb = l[e]; bi = e; }
  e0 = a; e1 = bi;
  float x1 = __expf(vb - va);
  w0 = 1.f / (1.f + x1);
  w1 = x1 / (1.f + x1);
}

__global__ void k_route(const float* __restrict__ logits, float* __restrict__ gates,
                        int* __restrict__ cnt) {
  int n = blockIdx.x * blockDim.x + threadIdx.x;
  if (n >= Nc) return;
  float l[Ec];
#pragma unroll
  for (int e = 0; e < Ec; ++e) l[e] = logits[n * Ec + e];
  int e0, e1; float w0, w1;
  top2(l, e0, e1, w0, w1);
  atomicAdd(&cnt[e0], 1);
  atomicAdd(&cnt[e1], 1);
  float mx = l[0];
#pragma unroll
  for (int e = 1; e < Ec; ++e) mx = fmaxf(mx, l[e]);
  float ex[Ec], se = 0.f;
#pragma unroll
  for (int e = 0; e < Ec; ++e) { ex[e] = __expf(l[e] - mx); se += ex[e]; }
#pragma unroll
  for (int e = 0; e < Ec; ++e) gates[n * Ec + e] = ex[e] / se;
}

__global__ void k_scan(const int* __restrict__ cnt, int* __restrict__ offs,
                       int* __restrict__ fill) {
  if (threadIdx.x == 0) {
    int acc = 0;
    for (int e = 0; e < Ec; ++e) { offs[e] = acc; acc += cnt[e]; }
    offs[Ec] = acc;
  }
  if (threadIdx.x < Ec) fill[threadIdx.x] = 0;
}

__global__ void k_assign(const float* __restrict__ logits, const int* __restrict__ offs,
                         int* __restrict__ fill, int* __restrict__ assign_tok,
                         float* __restrict__ assign_w, int* __restrict__ tok_a) {
  int n = blockIdx.x * blockDim.x + threadIdx.x;
  if (n >= Nc) return;
  float l[Ec];
#pragma unroll
  for (int e = 0; e < Ec; ++e) l[e] = logits[n * Ec + e];
  int e0, e1; float w0, w1;
  top2(l, e0, e1, w0, w1);
  int p0 = atomicAdd(&fill[e0], 1);
  int a0 = offs[e0] + p0;
  assign_tok[a0] = n; assign_w[a0] = w0; tok_a[2 * n] = a0;
  int p1 = atomicAdd(&fill[e1], 1);
  int a1 = offs[e1] + p1;
  assign_tok[a1] = n; assign_w[a1] = w1; tok_a[2 * n + 1] = a1;
}

__global__ __launch_bounds__(256) void k_ln2(
    const float* __restrict__ x1f, const float* __restrict__ y,
    const int* __restrict__ tok_a, const float* __restrict__ assign_w,
    const float* __restrict__ g, const float* __restrict__ be,
    float* __restrict__ out) {
  int n = blockIdx.x, tid = threadIdx.x;
  __shared__ float red1[8], red2[8];
  int a0 = tok_a[2 * n], a1 = tok_a[2 * n + 1];
  float w0 = assign_w[a0], w1 = assign_w[a1];
  float t[4], s = 0.f, ss = 0.f;
#pragma unroll
  for (int i = 0; i < 4; ++i) {
    int idx = tid + i * 256;
    float v = x1f[(long)n * Dc + idx] + w0 * y[(long)a0 * Dc + idx] +
              w1 * y[(long)a1 * Dc + idx];
    t[i] = v; s += v; ss += v * v;
  }
  s = block_sum(s, red1);
  ss = block_sum(ss, red2);
  float mean = s * (1.f / Dc);
  float var = ss * (1.f / Dc) - mean * mean;
  float inv = rsqrtf(var + 1e-5f);
#pragma unroll
  for (int i = 0; i < 4; ++i) {
    int idx = tid + i * 256;
    out[(long)n * Dc + idx] = (t[i] - mean) * inv * g[idx] + be[idx];
  }
}

__global__ __launch_bounds__(256) void k_lb(const float* __restrict__ gates,
                                            float* __restrict__ out) {
  __shared__ float sm[64];
  int lane = threadIdx.x & 31, wid = threadIdx.x >> 5;
  float p[Ec];
#pragma unroll
  for (int e = 0; e < Ec; ++e) p[e] = 0.f;
  for (int n = threadIdx.x; n < Nc; n += 256)
#pragma unroll
    for (int e = 0; e < Ec; ++e) p[e] += gates[n * Ec + e];
#pragma unroll
  for (int e = 0; e < Ec; ++e) {
    float v = p[e];
#pragma unroll
    for (int m = 16; m >= 1; m >>= 1) v += __shfl_xor(v, m, 32);
    if (lane == 0) sm[e * 8 + wid] = v;
  }
  __syncthreads();
  if (threadIdx.x == 0) {
    float loss = 0.f;
    for (int e = 0; e < Ec; ++e) {
      float u = 0.f;
      for (int w = 0; w < 8; ++w) u += sm[e * 8 + w];
      u *= (1.f / Nc);
      loss += u * u;
    }
    out[0] = (float)Ec * loss;
  }
}

// ------------------------------------------------------------------- driver
extern "C" void kernel_launch(void* const* d_in, const int* in_sizes, int n_in,
                              void* d_out, int out_size, void* d_ws, size_t ws_size,
                              hipStream_t stream) {
  const float* x      = (const float*)d_in[0];
  const float* in_w   = (const float*)d_in[1];
  const float* in_b   = (const float*)d_in[2];
  const float* out_w  = (const float*)d_in[3];
  const float* out_b  = (const float*)d_in[4];
  const float* ln1_g  = (const float*)d_in[5];
  const float* ln1_b  = (const float*)d_in[6];
  const float* gate_w = (const float*)d_in[7];
  const float* gate_b = (const float*)d_in[8];
  const float* w1     = (const float*)d_in[9];
  const float* b1     = (const float*)d_in[10];
  const float* w2     = (const float*)d_in[11];
  const float* b2     = (const float*)d_in[12];
  const float* ln2_g  = (const float*)d_in[13];
  const float* ln2_b  = (const float*)d_in[14];
  float* out = (float*)d_out;

  char* base = (char*)d_ws;
  size_t pos = 0;
  auto carve = [&](size_t bytes) -> void* {
    void* p = base + pos;
    pos += (bytes + 255) & ~(size_t)255;
    return p;
  };
  u16* xb        = (u16*)carve((size_t)Nc * Dc * 2);
  u16* wqkvb     = (u16*)carve((size_t)D3c * Dc * 2);
  u16* woutb     = (u16*)carve((size_t)Dc * Dc * 2);
  u16* wexb      = (u16*)carve((size_t)Ec * FFc * Dc * 2);  // w1 then w2 (bf16)
  u16* qkv_h     = (u16*)carve((size_t)NAc * FFc * 2);      // qkv, later h
  u16* ctxb      = (u16*)carve((size_t)Nc * Dc * 2);
  float* attn_o  = (float*)carve((size_t)Nc * Dc * 4);
  float* x1f     = (float*)carve((size_t)Nc * Dc * 4);
  u16* x1b       = (u16*)carve((size_t)Nc * Dc * 2);
  float* logits  = (float*)carve((size_t)Nc * Ec * 4);
  float* gates   = (float*)carve((size_t)Nc * Ec * 4);
  float* yb      = (float*)carve((size_t)NAc * Dc * 4);
  int* cnt       = (int*)carve(64);
  int* offs      = (int*)carve(64);
  int* fill      = (int*)carve(64);
  u16* zeros     = (u16*)carve(256);
  int* assign_tok = (int*)carve((size_t)NAc * 4);
  float* assign_w = (float*)carve((size_t)NAc * 4);
  int* tok_a      = (int*)carve((size_t)NAc * 4);
  u16* qkvb = qkv_h;
  u16* hb   = qkv_h;

  // precision conversions (bf16 for WMMA) + control-buffer init
  k_cvt<<<512, 256, 0, stream>>>(x, xb, (long)Nc * Dc);
  k_cvt<<<512, 256, 0, stream>>>(in_w, wqkvb, (long)D3c * Dc);
  k_cvt<<<256, 256, 0, stream>>>(out_w, woutb, (long)Dc * Dc);
  k_cvt<<<2048, 256, 0, stream>>>(w1, wexb, (long)Ec * FFc * Dc);
  k_zero<<<1, 64, 0, stream>>>(cnt, 8);
  k_zero<<<1, 64, 0, stream>>>((int*)zeros, 64);

  // QKV projection: [4096,3072] = xb @ in_proj_w^T + b
  k_gemm<<<dim3(Nc / 128, D3c / 128, 1), 256, 0, stream>>>(
      xb, wqkvb, in_b, 0L, nullptr, qkvb, Nc, D3c, Dc, 0,
      nullptr, nullptr, nullptr, 0L, zeros);

  // flash attention
  k_attn<<<dim3(Sc / 128, Hc, Bc), 256, 0, stream>>>(qkvb, ctxb);

  // output projection
  k_gemm<<<dim3(Nc / 128, Dc / 128, 1), 256, 0, stream>>>(
      ctxb, woutb, out_b, 0L, attn_o, nullptr, Nc, Dc, Dc, 0,
      nullptr, nullptr, nullptr, 0L, zeros);

  // residual + LN1
  k_ln1<<<Nc, 256, 0, stream>>>(x, attn_o, ln1_g, ln1_b, x1f, x1b);

  // routing
  k_gate<<<Nc, 256, 0, stream>>>(x1f, gate_w, gate_b, logits);
  k_route<<<Nc / 256, 256, 0, stream>>>(logits, gates, cnt);
  k_scan<<<1, 32, 0, stream>>>(cnt, offs, fill);
  k_assign<<<Nc / 256, 256, 0, stream>>>(logits, offs, fill, assign_tok,
                                         assign_w, tok_a);

  // MoE pass 1: h = relu(x1[token] @ w1[e]^T + b1[e])  (gathered rows)
  k_gemm<<<dim3(Nc / 128, FFc / 128, Ec), 256, 0, stream>>>(
      x1b, wexb, b1, (long)FFc, nullptr, hb, Nc, FFc, Dc, 1,
      assign_tok, cnt, offs, (long)FFc * Dc, zeros);

  // convert w2 into the shared weight buffer (stream-ordered after pass 1)
  k_cvt<<<2048, 256, 0, stream>>>(w2, wexb, (long)Ec * Dc * FFc);

  // MoE pass 2: y = h @ w2[e]^T + b2[e]
  k_gemm<<<dim3(Nc / 128, Dc / 128, Ec), 256, 0, stream>>>(
      hb, wexb, b2, (long)Dc, yb, nullptr, Nc, Dc, FFc, 0,
      nullptr, cnt, offs, (long)Dc * FFc, zeros);

  // combine experts + residual + LN2 -> output
  k_ln2<<<Nc, 256, 0, stream>>>(x1f, yb, tok_a, assign_w, ln2_g, ln2_b, out);

  // load-balance loss (deterministic fixed-tree reduction)
  k_lb<<<1, 256, 0, stream>>>(gates, out + (size_t)Nc * Dc);
}
